// KiperwasserDependencyParser_7447473291454
// MI455X (gfx1250) — matrix operand order repaired and checked by reading, compiled, not run on hardware
//
#include <hip/hip_runtime.h>
#include <stddef.h>
#include <stdint.h>

// ---------------------------------------------------------------------------
// Types for WMMA fragments (CDNA5 / gfx1250, wave32)
// ---------------------------------------------------------------------------
typedef __attribute__((ext_vector_type(16))) _Float16 v16h;
typedef __attribute__((ext_vector_type(8)))  _Float16 v8h;
typedef __attribute__((ext_vector_type(8)))  float    v8f;
typedef __attribute__((ext_vector_type(2)))  _Float16 h2;

#define NSEQ 512
#define HID  256
#define G4H  1024   // 4*H
#define D2H  512    // 2*H
#define MM   100    // MLP hidden
#define MMP  128    // padded to multiple of 64 (so a|b GEMM N = 256 = 4 wave-tiles)

// ---------------------------------------------------------------------------
// 1) Embedding gather + concat -> f16, K padded 125 -> 128
// ---------------------------------------------------------------------------
__global__ void embed_kernel(const int* __restrict__ widx, const int* __restrict__ pidx,
                             const float* __restrict__ wemb, const float* __restrict__ pemb,
                             _Float16* __restrict__ x0) {
    int idx = blockIdx.x * blockDim.x + threadIdx.x;      // 512*128
    if (idx >= NSEQ * 128) return;
    int t = idx >> 7, k = idx & 127;
    float v = 0.f;
    if (k < 100)      v = wemb[(size_t)widx[t] * 100 + k];
    else if (k < 125) v = pemb[(size_t)pidx[t] * 25 + (k - 100)];
    x0[idx] = (_Float16)v;
}

// ---------------------------------------------------------------------------
// 2) Generic f32 -> f16 row-pad convert: src [R][Cs] -> dst [R][Cd] (zero pad)
// ---------------------------------------------------------------------------
__global__ void cvt_pad_kernel(const float* __restrict__ src, _Float16* __restrict__ dst,
                               int R, int Cs, int Cd) {
    int idx = blockIdx.x * blockDim.x + threadIdx.x;
    if (idx >= R * Cd) return;
    int r = idx / Cd, c = idx % Cd;
    dst[idx] = (_Float16)((c < Cs) ? src[(size_t)r * Cs + c] : 0.f);
}

// ---------------------------------------------------------------------------
// 3) bias = bih + bhh (per direction, 2048 entries)
// ---------------------------------------------------------------------------
__global__ void bias_kernel(const float* __restrict__ bih, const float* __restrict__ bhh,
                            float* __restrict__ bias) {
    int i = blockIdx.x * blockDim.x + threadIdx.x;
    if (i < 2 * G4H) bias[i] = bih[i] + bhh[i];
}

// ---------------------------------------------------------------------------
// 4) Pack W1 [100][1024] -> [256][512] f16: rows 0..127 = head half, 128..255 = dep half
// ---------------------------------------------------------------------------
__global__ void packW1_kernel(const float* __restrict__ W1, _Float16* __restrict__ dst) {
    int idx = blockIdx.x * blockDim.x + threadIdx.x;      // 256*512
    if (idx >= 2 * MMP * D2H) return;
    int r = idx / D2H, c = idx % D2H;
    int m = r % MMP, part = r / MMP;
    float v = (m < MM) ? W1[(size_t)m * 1024 + part * D2H + c] : 0.f;
    dst[idx] = (_Float16)v;
}

// ---------------------------------------------------------------------------
// 5) WMMA GEMM:  C[M][N] = A[M][K](f16) * B[N][K](f16)^T  (+ bias[n])
//    One wave computes a 16x64 output strip (4 x 16x16 tiles). All fragment
//    loads (1x A, 4x B) are issued before the 4 v_wmma so the compiler can
//    clause them and stagger s_wait_loadcnt, overlapping VMEM with matrix ops.
//    Requires N % 64 == 0, K % 32 == 0.
//    Fragment layouts per CDNA5 ISA 7.12.2 (wave32).
// ---------------------------------------------------------------------------
#define NT 4
__global__ __launch_bounds__(32)
void wmma_gemm_kernel(const _Float16* __restrict__ A, const _Float16* __restrict__ B,
                      const float* __restrict__ bias, float* __restrict__ C,
                      int M, int N, int K) {
    int l   = threadIdx.x;        // 0..31
    int lan = l & 15;
    int hs  = l >> 4;             // half-wave select
    int n0  = blockIdx.x * (16 * NT);
    int m0  = blockIdx.y * 16;

    v8f acc[NT];
#pragma unroll
    for (int nt = 0; nt < NT; ++nt) acc[nt] = (v8f){};

    const _Float16* arow = A + (size_t)(m0 + lan) * K;    // A row for this lane (row = l%16)

    for (int k0 = 0; k0 < K; k0 += 32) {
        // ---- issue ALL fragment loads first (independent registers) ----
        // A fragment: lanes 0-15 K offsets {0..7,16..23}; lanes 16-31 {8..15,24..31}
        const _Float16* ap = arow + k0 + hs * 8;
        v8h a0 = *(const v8h*)(ap);
        v8h a1 = *(const v8h*)(ap + 16);
        // B fragments: col n = weight row n; lanes 0-15 K {0..7,16..23}; lanes 16-31 K {8..15,24..31}
        v8h b0[NT], b1[NT];
#pragma unroll
        for (int nt = 0; nt < NT; ++nt) {
            const _Float16* bp = B + (size_t)(n0 + nt * 16 + lan) * K + k0 + hs * 8;
            b0[nt] = *(const v8h*)(bp);
            b1[nt] = *(const v8h*)(bp + 16);
        }

        v16h af;
#pragma unroll
        for (int i = 0; i < 8; ++i) { af[i] = a0[i]; af[i + 8] = a1[i]; }

        // ---- 4 WMMAs sharing one A fragment ----
#pragma unroll
        for (int nt = 0; nt < NT; ++nt) {
            v16h bf;
#pragma unroll
            for (int i = 0; i < 8; ++i) { bf[i] = b0[nt][i]; bf[i + 8] = b1[nt][i]; }
            acc[nt] = __builtin_amdgcn_wmma_f32_16x16x32_f16(
                /*neg_a*/false, af, /*neg_b*/false, bf,
                /*c_mod*/(short)0, acc[nt], /*reuse_a*/false, /*reuse_b*/false);
        }
    }

#pragma unroll
    for (int nt = 0; nt < NT; ++nt) {
        int n = n0 + nt * 16 + lan;
        float bv = bias ? bias[n] : 0.f;
#pragma unroll
        for (int r = 0; r < 8; ++r) {
            int m = m0 + r + hs * 8;                      // C/D layout: VGPR r -> row r (+8 for hi half)
            C[(size_t)m * N + n] = acc[nt][r] + bv;
        }
    }
}

// ---------------------------------------------------------------------------
// 6) Persistent LSTM scan. One workgroup (1024 threads) per direction.
//    Thread t owns gate row t of Whh (f16): k=0..127 in 64 VGPR dwords,
//    k=128..255 in LDS (rotate-swizzled, conflict-free). h in LDS (f32).
//    Gate order i,f,g,o (PyTorch).
// ---------------------------------------------------------------------------
__device__ __forceinline__ float sigmf(float x) { return 1.f / (1.f + __expf(-x)); }

__global__ __launch_bounds__(1024)
void lstm_scan_kernel(const float* __restrict__ pre,    // [512][2048]
                      const float* __restrict__ Whh,    // [2][1024][256] f32
                      float* __restrict__ hseq) {       // [2][512][256]
    extern __shared__ char smem_raw[];
    unsigned* LW = (unsigned*)smem_raw;                   // 1024 rows * 64 dwords = 256 KB
    float*    sg = (float*)(smem_raw + 262144);           // 1024 f32 gates
    float*    sh = (float*)(smem_raw + 262144 + 4096);    // 256 f32 h

    const int d = blockIdx.x;                             // direction
    const int t = threadIdx.x;                            // gate row 0..1023
    const int dOff = d * G4H;

    // ---- load & convert this row of Whh ----
    const float* wr = Whh + ((size_t)d * G4H + t) * HID;
    unsigned wreg[64];
#pragma unroll
    for (int i = 0; i < 64; ++i) {
        h2 p; p.x = (_Float16)wr[2 * i]; p.y = (_Float16)wr[2 * i + 1];
        wreg[i] = __builtin_bit_cast(unsigned, p);
    }
    for (int p = 0; p < 64; ++p) {
        h2 q; q.x = (_Float16)wr[128 + 2 * p]; q.y = (_Float16)wr[129 + 2 * p];
        LW[(t << 6) + ((p + t) & 63)] = __builtin_bit_cast(unsigned, q);
    }
    if (t < HID) sh[t] = 0.f;
    float c = 0.f;
    __syncthreads();

    for (int s = 0; s < NSEQ; ++s) {
        const int tt = (d == 0) ? s : (NSEQ - 1 - s);     // reverse scan for dir 1
        float acc = pre[(size_t)tt * (2 * G4H) + dOff + t];
        // VGPR-resident half: k = 0..127  (f16 * f32 -> v_fma_mix)
#pragma unroll
        for (int i = 0; i < 64; ++i) {
            h2 wp = __builtin_bit_cast(h2, wreg[i]);
            acc += (float)wp.x * sh[2 * i];
            acc += (float)wp.y * sh[2 * i + 1];
        }
        // LDS-resident half: k = 128..255 (swizzled, bank-conflict free)
#pragma unroll 8
        for (int p = 0; p < 64; ++p) {
            h2 wp = __builtin_bit_cast(h2, LW[(t << 6) + ((p + t) & 63)]);
            acc += (float)wp.x * sh[128 + 2 * p];
            acc += (float)wp.y * sh[129 + 2 * p];
        }
        sg[t] = acc;
        __syncthreads();
        if (t < HID) {
            float gi = sg[t], gf = sg[HID + t], gg = sg[2 * HID + t], go = sg[3 * HID + t];
            c = sigmf(gf) * c + sigmf(gi) * tanhf(gg);
            float h = sigmf(go) * tanhf(c);
            sh[t] = h;
            hseq[((size_t)d * NSEQ + tt) * HID + t] = h;
        }
        __syncthreads();
    }
}

// ---------------------------------------------------------------------------
// 7) Concat fwd/bwd h sequences -> f16 [512][512]
// ---------------------------------------------------------------------------
__global__ void cat_kernel(const float* __restrict__ hseq, _Float16* __restrict__ dst) {
    int idx = blockIdx.x * blockDim.x + threadIdx.x;      // 512*512
    if (idx >= NSEQ * D2H) return;
    int t = idx >> 9, j = idx & 511;
    int dd = j >> 8, jj = j & 255;
    dst[idx] = (_Float16)hseq[((size_t)dd * NSEQ + t) * HID + jj];
}

// ---------------------------------------------------------------------------
// 8) Pairwise scorer: score[i][j] = b2 + sum_m W2[m]*tanh(a[i][m]+b[j][m]+b1[m])
//    ab layout: [512][256]  (cols 0..127 = a, 128..255 = b; only m<100 used)
// ---------------------------------------------------------------------------
__global__ __launch_bounds__(256)
void scorer_kernel(const float* __restrict__ ab, const float* __restrict__ b1,
                   const float* __restrict__ W2, const float* __restrict__ b2,
                   float* __restrict__ out) {
    __shared__ float sa[16][MMP];
    __shared__ float sb[16][MMP];
    __shared__ float sw[MMP];
    __shared__ float sbias[MMP];
    int tx = threadIdx.x, ty = threadIdx.y;
    int tid = ty * 16 + tx;
    int i0 = blockIdx.y * 16, j0 = blockIdx.x * 16;

    for (int idx = tid; idx < 16 * MMP; idx += 256) {
        int r = idx / MMP, cc = idx % MMP;
        sa[r][cc] = ab[(size_t)(i0 + r) * (2 * MMP) + cc];
        sb[r][cc] = ab[(size_t)(j0 + r) * (2 * MMP) + MMP + cc];
    }
    for (int idx = tid; idx < MMP; idx += 256) {
        sw[idx]    = (idx < MM) ? W2[idx] : 0.f;
        sbias[idx] = (idx < MM) ? b1[idx] : 0.f;
    }
    __syncthreads();

    float acc = b2[0];
#pragma unroll 4
    for (int m = 0; m < MM; ++m)
        acc += sw[m] * tanhf(sa[ty][m] + sb[tx][m] + sbias[m]);
    out[(size_t)(i0 + ty) * NSEQ + (j0 + tx)] = acc;
}

// ---------------------------------------------------------------------------
// Host-side launch
// ---------------------------------------------------------------------------
extern "C" void kernel_launch(void* const* d_in, const int* in_sizes, int n_in,
                              void* d_out, int out_size, void* d_ws, size_t ws_size,
                              hipStream_t stream) {
    const int*   widx = (const int*)d_in[0];
    const int*   pidx = (const int*)d_in[1];
    const float* wemb = (const float*)d_in[2];
    const float* pemb = (const float*)d_in[3];
    const float* Wih0 = (const float*)d_in[4];
    const float* Whh0 = (const float*)d_in[5];
    const float* bih0 = (const float*)d_in[6];
    const float* bhh0 = (const float*)d_in[7];
    const float* Wih1 = (const float*)d_in[8];
    const float* Whh1 = (const float*)d_in[9];
    const float* bih1 = (const float*)d_in[10];
    const float* bhh1 = (const float*)d_in[11];
    const float* W1   = (const float*)d_in[12];
    const float* b1   = (const float*)d_in[13];
    const float* W2   = (const float*)d_in[14];
    const float* b2   = (const float*)d_in[15];
    float* scores = (float*)d_out;

    // carve workspace (256B-aligned)
    char* w = (char*)d_ws; size_t off = 0;
    auto alloc = [&](size_t bytes) -> void* {
        void* p = w + off; off = (off + bytes + 255) & ~(size_t)255; return p;
    };
    _Float16* x0f16   = (_Float16*)alloc((size_t)NSEQ * 128 * 2);
    _Float16* wih0f16 = (_Float16*)alloc((size_t)2 * G4H * 128 * 2);
    _Float16* wih1f16 = (_Float16*)alloc((size_t)2 * G4H * D2H * 2);
    _Float16* w1f16   = (_Float16*)alloc((size_t)2 * MMP * D2H * 2);
    float*    bias0   = (float*)alloc((size_t)2 * G4H * 4);
    float*    bias1   = (float*)alloc((size_t)2 * G4H * 4);
    float*    pre0    = (float*)alloc((size_t)NSEQ * 2 * G4H * 4);
    float*    pre1    = (float*)alloc((size_t)NSEQ * 2 * G4H * 4);
    float*    h0seq   = (float*)alloc((size_t)2 * NSEQ * HID * 4);
    float*    h1seq   = (float*)alloc((size_t)2 * NSEQ * HID * 4);
    _Float16* h0cat   = (_Float16*)alloc((size_t)NSEQ * D2H * 2);
    _Float16* h1cat   = (_Float16*)alloc((size_t)NSEQ * D2H * 2);
    float*    ab      = (float*)alloc((size_t)NSEQ * 2 * MMP * 4);
    (void)ws_size; (void)n_in; (void)in_sizes; (void)out_size;

    // 1) embeddings + weight preparation
    embed_kernel<<<(NSEQ * 128 + 255) / 256, 256, 0, stream>>>(widx, pidx, wemb, pemb, x0f16);
    cvt_pad_kernel<<<(2 * G4H * 128 + 255) / 256, 256, 0, stream>>>(Wih0, wih0f16, 2 * G4H, 125, 128);
    cvt_pad_kernel<<<(2 * G4H * D2H + 255) / 256, 256, 0, stream>>>(Wih1, wih1f16, 2 * G4H, D2H, D2H);
    packW1_kernel<<<(2 * MMP * D2H + 255) / 256, 256, 0, stream>>>(W1, w1f16);
    bias_kernel<<<(2 * G4H + 255) / 256, 256, 0, stream>>>(bih0, bhh0, bias0);
    bias_kernel<<<(2 * G4H + 255) / 256, 256, 0, stream>>>(bih1, bhh1, bias1);

    // 2) pre0 = x0 @ Wih0^T + bias0   (M=512, N=2048, K=128)
    wmma_gemm_kernel<<<dim3(2 * G4H / 64, NSEQ / 16), 32, 0, stream>>>(
        x0f16, wih0f16, bias0, pre0, NSEQ, 2 * G4H, 128);

    // 3) layer-0 scan (fwd + bwd workgroups), then concat to f16
    size_t scan_smem = 262144 + 4096 + 1024;
    lstm_scan_kernel<<<2, 1024, scan_smem, stream>>>(pre0, Whh0, h0seq);
    cat_kernel<<<(NSEQ * D2H + 255) / 256, 256, 0, stream>>>(h0seq, h0cat);

    // 4) pre1 = h0cat @ Wih1^T + bias1  (M=512, N=2048, K=512)
    wmma_gemm_kernel<<<dim3(2 * G4H / 64, NSEQ / 16), 32, 0, stream>>>(
        h0cat, wih1f16, bias1, pre1, NSEQ, 2 * G4H, D2H);

    // 5) layer-1 scan, concat
    lstm_scan_kernel<<<2, 1024, scan_smem, stream>>>(pre1, Whh1, h1seq);
    cat_kernel<<<(NSEQ * D2H + 255) / 256, 256, 0, stream>>>(h1seq, h1cat);

    // 6) [a|b] = h1cat @ [W1a;W1b]^T  (M=512, N=256, K=512), no bias
    wmma_gemm_kernel<<<dim3(2 * MMP / 64, NSEQ / 16), 32, 0, stream>>>(
        h1cat, w1f16, (const float*)nullptr, ab, NSEQ, 2 * MMP, D2H);

    // 7) pairwise scorer -> scores [512][512]
    scorer_kernel<<<dim3(NSEQ / 16, NSEQ / 16), dim3(16, 16), 0, stream>>>(ab, b1, W2, b2, scores);
}
